// GNN_Layer_72834055406175
// MI455X (gfx1250) — compile-verified, matrix operand order checked by reading, not run on hardware
//
#include <hip/hip_runtime.h>
#include <hip/hip_bf16.h>

// GNN layer for MI455X (gfx1250): bf16 WMMA 16x16x32 with f32 accumulate,
// TDM (tensor_load_to_lds) staging of the pre-transposed bf16 x operand.
// N = 8192 rows, HIDDEN = OUT = 512.

typedef __attribute__((ext_vector_type(16))) __bf16 v16bf;
typedef __attribute__((ext_vector_type(8)))  float  v8f;
typedef unsigned int u32x4 __attribute__((ext_vector_type(4)));
typedef int          i32x8 __attribute__((ext_vector_type(8)));
typedef int          i32x4 __attribute__((ext_vector_type(4)));

#define NROWS 8192
#define HID   512

#if defined(__has_builtin)
#if __has_builtin(__builtin_amdgcn_tensor_load_to_lds) && __has_builtin(__builtin_amdgcn_s_wait_tensorcnt)
#define USE_TDM 1
#endif
#endif
#ifndef USE_TDM
#define USE_TDM 0
#endif

// ---------------------------------------------------------------------------
// Pre-pass: xT[n][k] = (bf16) x[k][n]   (x: [8192, 512] f32 -> xT: [512, 8192] bf16)
// ---------------------------------------------------------------------------
__global__ __launch_bounds__(256)
void transpose_convert_kernel(const float* __restrict__ x, __bf16* __restrict__ xT)
{
    __shared__ float tile[32][33];
    const int k0 = blockIdx.x * 32;       // row in x (0..8191)
    const int n0 = blockIdx.y * 32;       // col in x (0..511)
    const int tx = threadIdx.x;           // 0..31
    const int ty = threadIdx.y;           // 0..7
    #pragma unroll
    for (int r = ty; r < 32; r += 8)
        tile[r][tx] = x[(size_t)(k0 + r) * HID + n0 + tx];
    __syncthreads();
    #pragma unroll
    for (int r = ty; r < 32; r += 8)
        xT[(size_t)(n0 + r) * NROWS + k0 + tx] = (__bf16)tile[tx][r];
}

// ---------------------------------------------------------------------------
// Aggregation GEMM: C[8192,512] = A[8192,8192](f32) @ x, with B = xT (bf16).
// Tile BM=128 x BN=128 x BK=32; 256 threads = 8 wave32; 16x128 strip per wave
// (8 accumulators, 8 WMMAs per A-fragment). B tile staged by the Tensor Data
// Mover directly into LDS (bf16, no conversion); A tile converted on stage.
// ---------------------------------------------------------------------------
__global__ __launch_bounds__(256)
void agg_gemm_kernel(const float* __restrict__ Ag, const __bf16* __restrict__ xT,
                     float* __restrict__ C)
{
    __shared__ __bf16 As[128][32];   // 8 KB
    __shared__ __bf16 Bt[128][32];   // 8 KB, Bt[n][k] = x[k][nBlk+n]

    const int t    = threadIdx.x;
    const int lane = t & 31;
    const int wv   = t >> 5;
    const int mBlk = blockIdx.y * 128;
    const int nBlk = blockIdx.x * 128;
    const int m0w  = wv * 16;

    v8f acc[8] = {};

    const unsigned ldsB = (unsigned)(uintptr_t)&Bt[0][0];           // LDS byte offset
    const unsigned long long gB =
        (unsigned long long)(uintptr_t)(xT + (size_t)nBlk * NROWS); // tile row 0 addr

    for (int kt = 0; kt < NROWS; kt += 32) {
#if USE_TDM
        if (wv == 0) {
            // ---- Tensor DMA descriptor (D#), §8.3/8.4: 2D bf16 tile 32(k) x 128(n)
            const unsigned long long ga = gB + (unsigned long long)kt * 2;
            u32x4 g0;
            g0[0] = 1u;                                            // count=1 (user)
            g0[1] = ldsB;                                          // lds_addr
            g0[2] = (unsigned)(ga & 0xFFFFFFFFu);                  // global_addr[31:0]
            g0[3] = (unsigned)((ga >> 32) & 0x01FFFFFFu) | (2u << 30); // addr[56:32], type=2
            i32x8 g1;
            g1[0] = (int)(1u << 16);                               // data_size=1 (2 bytes)
            g1[1] = (int)(((unsigned)NROWS & 0xFFFFu) << 16);      // tensor_dim0[15:0]=8192
            g1[2] = (int)(((unsigned)NROWS >> 16) |
                          (((unsigned)HID & 0xFFFFu) << 16));      // dim0 hi | dim1 lo (512)
            g1[3] = (int)(((unsigned)HID >> 16) | (32u << 16));    // dim1 hi | tile_dim0=32
            g1[4] = 128;                                           // tile_dim1=128, tile_dim2=0
            g1[5] = NROWS;                                         // dim0_stride lo = 8192
            g1[6] = 0;                                             // stride hi | dim1_stride lo
            g1[7] = 0;
            i32x4 g2 = {0, 0, 0, 0};
            i32x4 g3 = {0, 0, 0, 0};
            i32x8 g4 = {0, 0, 0, 0, 0, 0, 0, 0};
            __builtin_amdgcn_tensor_load_to_lds(g0, g1, g2, g3, g4, 0);
        }
#else
        // Fallback: manual bf16 copy of B tile (no conversion needed)
        #pragma unroll
        for (int r = 0; r < 2; ++r) {
            int idx = t + r * 256;                 // 0..511 16-byte chunks
            int row = idx >> 2;                    // 4 chunks per 64 B row
            int off = (idx & 3) << 3;              // bf16 elements
            const uint4 v = *(const uint4*)(xT + (size_t)(nBlk + row) * NROWS + kt + off);
            *(uint4*)&Bt[row][off] = v;
        }
#endif
        // ---- stage A tile: 128x32 f32 -> bf16 LDS (coalesced b128 loads)
        #pragma unroll
        for (int r = 0; r < 4; ++r) {
            int idx = t + r * 256;
            int row = idx >> 3;
            int kq  = (idx & 7) << 2;
            const float4 v = *(const float4*)(Ag + (size_t)(mBlk + row) * NROWS + kt + kq);
            As[row][kq + 0] = (__bf16)v.x;
            As[row][kq + 1] = (__bf16)v.y;
            As[row][kq + 2] = (__bf16)v.z;
            As[row][kq + 3] = (__bf16)v.w;
            if (r == 0) // speculative prefetch of next K-slab
                __builtin_prefetch(Ag + (size_t)(mBlk + row) * NROWS + kt + 32, 0, 0);
        }
#if USE_TDM
        if (wv == 0) __builtin_amdgcn_s_wait_tensorcnt(0);
#endif
        __syncthreads();

        // ---- A fragment (lanes 0-15: K{0-7,16-23}; lanes 16-31: K{8-15,24-31})
        v16bf af;
        const int mr = m0w + (lane & 15);
        const int kb = (lane >> 4) << 3;
        #pragma unroll
        for (int i = 0; i < 8; ++i) {
            int k0 = ((i < 4) ? (2 * i) : (8 + 2 * i)) + kb;
            af[2 * i]     = As[mr][k0];
            af[2 * i + 1] = As[mr][k0 + 1];
        }
        // ---- 8 WMMAs across the 128-col strip (A-fragment reused 8x)
        #pragma unroll
        for (int nb = 0; nb < 8; ++nb) {
            v16bf bfr;
            const int nc = nb * 16 + (lane & 15);
            #pragma unroll
            for (int i = 0; i < 8; ++i) {
                int k0 = ((i < 4) ? (2 * i) : (8 + 2 * i)) + kb;
                bfr[2 * i]     = Bt[nc][k0];
                bfr[2 * i + 1] = Bt[nc][k0 + 1];
            }
            acc[nb] = __builtin_amdgcn_wmma_f32_16x16x32_bf16(
                false, af, false, bfr, (short)0, acc[nb], false, false);
        }
        __syncthreads();
    }

    // ---- epilogue: plain f32 store
    #pragma unroll
    for (int nb = 0; nb < 8; ++nb) {
        const int col = nBlk + nb * 16 + (lane & 15);
        #pragma unroll
        for (int r = 0; r < 8; ++r) {
            const int row = mBlk + m0w + r + ((lane >> 4) << 3);
            C[(size_t)row * HID + col] = acc[nb][r];
        }
    }
}

// ---------------------------------------------------------------------------
// Fused small GEMM: C[M,512] = sum_p A_p[M,512] * W_p^T (+ biases/extra, relu)
//   W_p is [512,512] row-major (out-major) -> k contiguous, staged direct.
//   POOL: epilogue accumulates sum_rows mask[row]*C[row,col] into hg_accum.
//   Tile BM=128 x BN=64 x BK=32; 256 threads = 8 wave32.
// ---------------------------------------------------------------------------
template<int NPAIRS, bool RELU, bool POOL, bool WRITE_C>
__global__ __launch_bounds__(256)
void wmma_gemm_kernel(const float* __restrict__ A0, const float* __restrict__ A1,
                      const float* __restrict__ A2,
                      const float* __restrict__ B0, const float* __restrict__ B1,
                      const float* __restrict__ B2,
                      const float* __restrict__ bias0, const float* __restrict__ bias1,
                      const float* __restrict__ bias2,
                      const float* __restrict__ extra, const float* __restrict__ mask,
                      float* __restrict__ hg_accum, float* __restrict__ C)
{
    __shared__ __bf16 As[128][32];
    __shared__ __bf16 Bt[64][32];
    __shared__ float  pool[64];

    const int t    = threadIdx.x;
    const int lane = t & 31;
    const int wv   = t >> 5;
    const int mBlk = blockIdx.y * 128;
    const int nBlk = blockIdx.x * 64;
    const int m0w  = wv * 16;
    const int K    = HID;

    v8f acc[4] = {};

    const float* Ap[3] = {A0, A1, A2};
    const float* Bp[3] = {B0, B1, B2};

    #pragma unroll
    for (int p = 0; p < NPAIRS; ++p) {
        const float* __restrict__ Ag = Ap[p];
        const float* __restrict__ Bg = Bp[p];

        for (int kt = 0; kt < K; kt += 32) {
            #pragma unroll
            for (int r = 0; r < 4; ++r) {
                int idx = t + r * 256;
                int row = idx >> 3;
                int kq  = (idx & 7) << 2;
                const float4 v = *(const float4*)(Ag + (size_t)(mBlk + row) * K + kt + kq);
                As[row][kq + 0] = (__bf16)v.x;
                As[row][kq + 1] = (__bf16)v.y;
                As[row][kq + 2] = (__bf16)v.z;
                As[row][kq + 3] = (__bf16)v.w;
            }
            #pragma unroll
            for (int r = 0; r < 2; ++r) {
                int idx = t + r * 256;
                int n   = idx >> 3;
                int kq  = (idx & 7) << 2;
                const float4 v = *(const float4*)(Bg + (size_t)(nBlk + n) * K + kt + kq);
                Bt[n][kq + 0] = (__bf16)v.x;
                Bt[n][kq + 1] = (__bf16)v.y;
                Bt[n][kq + 2] = (__bf16)v.z;
                Bt[n][kq + 3] = (__bf16)v.w;
            }
            __syncthreads();

            v16bf af;
            const int mr = m0w + (lane & 15);
            const int kb = (lane >> 4) << 3;
            #pragma unroll
            for (int i = 0; i < 8; ++i) {
                int k0 = ((i < 4) ? (2 * i) : (8 + 2 * i)) + kb;
                af[2 * i]     = As[mr][k0];
                af[2 * i + 1] = As[mr][k0 + 1];
            }
            #pragma unroll
            for (int nb = 0; nb < 4; ++nb) {
                v16bf bfr;
                const int nc = nb * 16 + (lane & 15);
                #pragma unroll
                for (int i = 0; i < 8; ++i) {
                    int k0 = ((i < 4) ? (2 * i) : (8 + 2 * i)) + kb;
                    bfr[2 * i]     = Bt[nc][k0];
                    bfr[2 * i + 1] = Bt[nc][k0 + 1];
                }
                acc[nb] = __builtin_amdgcn_wmma_f32_16x16x32_bf16(
                    false, af, false, bfr, (short)0, acc[nb], false, false);
            }
            __syncthreads();
        }
    }

    if (POOL) {
        if (t < 64) pool[t] = 0.f;
        __syncthreads();
    }

    #pragma unroll
    for (int nb = 0; nb < 4; ++nb) {
        const int col = nBlk + nb * 16 + (lane & 15);
        float badd = 0.f;
        if (bias0) badd += bias0[col];
        if (NPAIRS > 1 && bias1) badd += bias1[col];
        if (NPAIRS > 2 && bias2) badd += bias2[col];
        if (extra) badd += extra[col];
        float psum = 0.f;
        #pragma unroll
        for (int r = 0; r < 8; ++r) {
            const int row = mBlk + m0w + r + ((lane >> 4) << 3);
            float v = acc[nb][r] + badd;
            if (RELU) v = fmaxf(v, 0.f);
            if (WRITE_C) C[(size_t)row * HID + col] = v;
            if (POOL) psum += mask[row] * v;
        }
        if (POOL) atomicAdd(&pool[nb * 16 + (lane & 15)], psum); // ds_add_f32
    }

    if (POOL) {
        __syncthreads();
        if (t < 64) atomicAdd(&hg_accum[nBlk + t], pool[t]);     // global_atomic_add_f32
    }
}

// ---------------------------------------------------------------------------
__global__ void zero_kernel(float* p, int n) {
    int i = blockIdx.x * blockDim.x + threadIdx.x;
    if (i < n) p[i] = 0.f;
}

// h_g = hg_accum / sum(mask); write h_g output; hgo = h_g @ W_go^T + b_go
__global__ __launch_bounds__(256)
void pool_finalize_kernel(const float* __restrict__ mask,
                          const float* __restrict__ hg_accum,
                          const float* __restrict__ W_go,
                          const float* __restrict__ b_go,
                          float* __restrict__ hgo,
                          float* __restrict__ hg_out)
{
    __shared__ float hg[HID];
    __shared__ float red[256];
    const int t = threadIdx.x;

    float s = 0.f;
    for (int i = t; i < NROWS; i += 256) s += mask[i];
    red[t] = s;
    __syncthreads();
    for (int off = 128; off > 0; off >>= 1) {
        if (t < off) red[t] += red[t + off];
        __syncthreads();
    }
    const float nsum = red[0];

    for (int o = t; o < HID; o += 256) {
        float v = hg_accum[o] / nsum;
        hg[o] = v;
        hg_out[o] = v;
    }
    __syncthreads();

    for (int o = t; o < HID; o += 256) {
        float a = b_go[o];
        const float* w = W_go + (size_t)o * HID;
        for (int j = 0; j < HID; ++j) a += hg[j] * w[j];
        hgo[o] = a;
    }
}

// ---------------------------------------------------------------------------
extern "C" void kernel_launch(void* const* d_in, const int* in_sizes, int n_in,
                              void* d_out, int out_size, void* d_ws, size_t ws_size,
                              hipStream_t stream)
{
    const float* x    = (const float*)d_in[0];
    const float* mask = (const float*)d_in[1];
    const float* a_ud = (const float*)d_in[2];
    const float* a_lr = (const float*)d_in[3];
    const float* W    = (const float*)d_in[4];
    const float* b    = (const float*)d_in[5];
    const float* W_ud = (const float*)d_in[6];
    const float* b_ud = (const float*)d_in[7];
    const float* W_lr = (const float*)d_in[8];
    const float* b_lr = (const float*)d_in[9];
    const float* W_g  = (const float*)d_in[10];
    const float* b_g  = (const float*)d_in[11];
    const float* W_go = (const float*)d_in[12];
    const float* b_go = (const float*)d_in[13];

    float* out_h  = (float*)d_out;                       // [8192, 512]
    float* out_hg = out_h + (size_t)NROWS * HID;         // [512]

    float*  ws       = (float*)d_ws;
    float*  h_ud     = ws;                               // [8192, 512] f32
    float*  h_lr     = h_ud + (size_t)NROWS * HID;       // [8192, 512] f32
    float*  hg_accum = h_lr + (size_t)NROWS * HID;       // [512]
    float*  hgo      = hg_accum + HID;                   // [512]
    __bf16* xT       = (__bf16*)(hgo + HID);             // [512, 8192] bf16

    zero_kernel<<<dim3(2), dim3(256), 0, stream>>>(hg_accum, HID);

    // x -> xT (transpose + f32->bf16), TDM-friendly B operand for aggregation
    transpose_convert_kernel<<<dim3(NROWS / 32, HID / 32), dim3(32, 8), 0, stream>>>(x, xT);

    dim3 blk(256);
    dim3 gridAgg(HID / 128, NROWS / 128);                // (4, 64)

    // h_ud = a_ud @ x ; h_lr = a_lr @ x   (K = 8192, B staged by TDM)
    agg_gemm_kernel<<<gridAgg, blk, 0, stream>>>(a_ud, xT, h_ud);
    agg_gemm_kernel<<<gridAgg, blk, 0, stream>>>(a_lr, xT, h_lr);

    dim3 gridSm(HID / 64, NROWS / 128);                  // (8, 64)

    // masked pool of relu(x @ W_g^T + b_g): fused in epilogue, no G array
    wmma_gemm_kernel<1, true, true, false><<<gridSm, blk, 0, stream>>>(
        x, nullptr, nullptr, W_g, nullptr, nullptr,
        b_g, nullptr, nullptr, nullptr, mask, hg_accum, nullptr);

    // h_g = accum / sum(mask);  hgo = h_g @ W_go^T + b_go;  write h_g output
    pool_finalize_kernel<<<dim3(1), dim3(256), 0, stream>>>(
        mask, hg_accum, W_go, b_go, hgo, out_hg);

    // h = relu(x@W^T + h_ud@W_ud^T + h_lr@W_lr^T + b + b_ud + b_lr + hgo)
    wmma_gemm_kernel<3, true, false, true><<<gridSm, blk, 0, stream>>>(
        x, h_ud, h_lr, W, W_ud, W_lr,
        b, b_ud, b_lr, hgo, nullptr, nullptr, out_h);
}
